// EdgeGATConv_81372450390256
// MI455X (gfx1250) — compile-verified, matrix operand order checked by reading, not run on hardware
//
#include <hip/hip_runtime.h>
#include <hip/hip_bf16.h>

typedef __attribute__((ext_vector_type(16))) _Float16 v16h;
typedef __attribute__((ext_vector_type(8)))  _Float16 v8h;
typedef __attribute__((ext_vector_type(8)))  float    v8f;

__device__ __forceinline__ float lrelu(float x) { return x >= 0.f ? x : 0.2f * x; }

__device__ __forceinline__ void atomicMaxF(float* addr, float v) {
    // monotone-in-both-encodings float max via int/uint atomics
    if (v >= 0.f) atomicMax((int*)addr, __float_as_int(v));
    else          atomicMin((unsigned int*)addr, (unsigned int)__float_as_int(v));
}

// ---- CDNA5 async global->LDS copy (ASYNCcnt-tracked, no VGPR data path) ----
__device__ __forceinline__ void async_g2l_b128(unsigned int lds_addr, const void* gptr) {
    asm volatile("global_load_async_to_lds_b128 %0, %1, off"
                 :: "v"(lds_addr), "v"(gptr)
                 : "memory");
}
__device__ __forceinline__ void wait_asynccnt0() {
    asm volatile("s_wait_asynccnt 0x0" ::: "memory");
}

// ---- WMMA operand loaders --------------------------------------------------
// A (16x32 f16): lane half h holds K in [k0+8h, k0+8h+8) then [k0+16+8h, ...)
__device__ __forceinline__ v16h load_a_f32row(const float* __restrict__ row, int k0, int half_) {
    const float4* p0 = (const float4*)(row + k0 + half_ * 8);
    const float4* p1 = (const float4*)(row + k0 + 16 + half_ * 8);
    float4 x0 = p0[0], x1 = p0[1], x2 = p1[0], x3 = p1[1];
    v16h a;
    a[0]  = (_Float16)x0.x; a[1]  = (_Float16)x0.y; a[2]  = (_Float16)x0.z; a[3]  = (_Float16)x0.w;
    a[4]  = (_Float16)x1.x; a[5]  = (_Float16)x1.y; a[6]  = (_Float16)x1.z; a[7]  = (_Float16)x1.w;
    a[8]  = (_Float16)x2.x; a[9]  = (_Float16)x2.y; a[10] = (_Float16)x2.z; a[11] = (_Float16)x2.w;
    a[12] = (_Float16)x3.x; a[13] = (_Float16)x3.y; a[14] = (_Float16)x3.z; a[15] = (_Float16)x3.w;
    return a;
}

// B (32x16 f16) from LDS-staged weights, row n = output column (C = X * W^T)
__device__ __forceinline__ v16h load_b_lds(const _Float16* __restrict__ Wlds, int n, int k0, int half_) {
    const v8h* q0 = (const v8h*)(Wlds + n * 128 + k0 + half_ * 8);
    const v8h* q1 = (const v8h*)(Wlds + n * 128 + k0 + 16 + half_ * 8);
    v8h b0 = *q0, b1 = *q1;
    v16h b;
#pragma unroll
    for (int j = 0; j < 8; ++j) { b[j] = b0[j]; b[8 + j] = b1[j]; }
    return b;
}

// ---- one-time f32 -> f16 weight conversion (128x128 matrix) ----------------
__global__ void w_to_f16_kernel(const float* __restrict__ W, _Float16* __restrict__ Wh) {
    int i = blockIdx.x * blockDim.x + threadIdx.x;  // float4 chunks
    if (i < 4096) {
        float4 w = ((const float4*)W)[i];
        Wh[4 * i + 0] = (_Float16)w.x;
        Wh[4 * i + 1] = (_Float16)w.y;
        Wh[4 * i + 2] = (_Float16)w.z;
        Wh[4 * i + 3] = (_Float16)w.w;
    }
}

// ---- stage 32KB of f16 weights into LDS via async copies -------------------
__device__ __forceinline__ void stage_weights_async(_Float16* Wlds, const _Float16* __restrict__ Wh,
                                                    int tid) {
    unsigned int lds0 = (unsigned int)(uintptr_t)(&Wlds[0]);
#pragma unroll
    for (int j = 0; j < 8; ++j) {
        int chunk = tid + j * 256;                      // 16-byte chunks
        async_g2l_b128(lds0 + chunk * 16, (const void*)(Wh + chunk * 8));
    }
    wait_asynccnt0();
    __syncthreads();
}

// ---- init: zero agg, -inf maxes, zero sums ---------------------------------
__global__ void init_kernel(float* __restrict__ agg, float* __restrict__ m1, float* __restrict__ m2,
                            float* __restrict__ s1, float* __restrict__ s2, int n_nodes) {
    int idx = blockIdx.x * blockDim.x + threadIdx.x;
    if (idx < n_nodes * 128) agg[idx] = 0.f;
    if (idx < n_nodes * 4) {
        m1[idx] = -__builtin_inff();
        m2[idx] = -__builtin_inff();
        s1[idx] = 0.f;
        s2[idx] = 0.f;
    }
}

// ---- H = X @ W^T via v_wmma_f32_16x16x32_f16, W(f16) async-staged in LDS ---
__global__ __launch_bounds__(256) void gemm_store_kernel(const float* __restrict__ Xin,
                                                         const _Float16* __restrict__ Wh,
                                                         float* __restrict__ Hout, int n_rows) {
    __shared__ _Float16 Wlds[128 * 128];  // 32 KB
    int tid = threadIdx.x;
    stage_weights_async(Wlds, Wh, tid);

    int wave = tid >> 5, lane = tid & 31;
    int half_ = lane >> 4, lr = lane & 15;
    long r0w = (long)blockIdx.x * 128 + wave * 16;
    long lrow = r0w + lr;
    if (lrow >= n_rows) lrow = n_rows - 1;  // clamp loads; keep EXEC full for WMMA
    const float* xrow = Xin + lrow * 128;

    v8f acc[8];
#pragma unroll
    for (int nt = 0; nt < 8; ++nt)
#pragma unroll
        for (int i = 0; i < 8; ++i) acc[nt][i] = 0.f;

#pragma unroll
    for (int ks = 0; ks < 4; ++ks) {
        int k0 = ks * 32;
        v16h a = load_a_f32row(xrow, k0, half_);
#pragma unroll
        for (int nt = 0; nt < 8; ++nt) {
            v16h b = load_b_lds(Wlds, nt * 16 + lr, k0, half_);
            acc[nt] = __builtin_amdgcn_wmma_f32_16x16x32_f16(false, a, false, b,
                                                             (short)0, acc[nt], false, false);
        }
    }
    // C/D layout: element i of lane -> (M = half_*8 + i, N = nt*16 + lr)
#pragma unroll
    for (int nt = 0; nt < 8; ++nt) {
        int n = nt * 16 + lr;
#pragma unroll
        for (int i = 0; i < 8; ++i) {
            long m = r0w + half_ * 8 + i;
            if (m < n_rows) Hout[m * 128 + n] = acc[nt][i];
        }
    }
}

// ---- per-node attention logits from stored h_src/h_dst (wave per node) -----
__global__ void e_node_kernel(const float* __restrict__ h_src, const float* __restrict__ h_dst,
                              const float* __restrict__ attn_src, const float* __restrict__ attn_dst,
                              float* __restrict__ e_src, float* __restrict__ e_dst, int n_nodes) {
    int gid = blockIdx.x * blockDim.x + threadIdx.x;
    int node = gid >> 5, lane = gid & 31;
    if (node >= n_nodes) return;
    int c0 = lane * 4, h = lane >> 3, d0 = (lane & 7) * 4;
    float4 hs = *(const float4*)(h_src + (size_t)node * 128 + c0);
    float4 hd = *(const float4*)(h_dst + (size_t)node * 128 + c0);
    float4 as = *(const float4*)(attn_src + h * 32 + d0);
    float4 ad = *(const float4*)(attn_dst + h * 32 + d0);
    float ps = hs.x * as.x + hs.y * as.y + hs.z * as.z + hs.w * as.w;
    float pd = hd.x * ad.x + hd.y * ad.y + hd.z * ad.z + hd.w * ad.w;
#pragma unroll
    for (int m = 1; m < 8; m <<= 1) {
        ps += __shfl_xor(ps, m, 32);
        pd += __shfl_xor(pd, m, 32);
    }
    if ((lane & 7) == 0) {
        e_src[node * 4 + h] = ps;
        e_dst[node * 4 + h] = pd;
    }
}

// ---- u[h][k] = sum_d attn_edge[h][d] * W_edge[h*32+d][k] -------------------
__global__ void u_edge_kernel(const float* __restrict__ attn_edge, const float* __restrict__ W_edge,
                              float* __restrict__ U) {
    int idx = blockIdx.x * blockDim.x + threadIdx.x;
    if (idx >= 512) return;
    int h = idx >> 7, k = idx & 127;
    float s = 0.f;
#pragma unroll 8
    for (int d = 0; d < 32; ++d) s += attn_edge[h * 32 + d] * W_edge[(size_t)(h * 32 + d) * 128 + k];
    U[idx] = s;
}

// ---- e_edge[e][h] = X_edge[e] . u[h]  (wave per edge, coalesced row read) --
__global__ void e_edge_kernel(const float* __restrict__ Xe, const float* __restrict__ U,
                              float* __restrict__ e_edge, int n_edges) {
    int gid = blockIdx.x * blockDim.x + threadIdx.x;
    int e = gid >> 5, lane = gid & 31;
    if (e >= n_edges) return;
    float4 x = *(const float4*)(Xe + (size_t)e * 128 + lane * 4);
    float p0, p1, p2, p3;
    {
        float4 u0 = *(const float4*)(U + 0 * 128 + lane * 4);
        float4 u1 = *(const float4*)(U + 1 * 128 + lane * 4);
        float4 u2 = *(const float4*)(U + 2 * 128 + lane * 4);
        float4 u3 = *(const float4*)(U + 3 * 128 + lane * 4);
        p0 = x.x * u0.x + x.y * u0.y + x.z * u0.z + x.w * u0.w;
        p1 = x.x * u1.x + x.y * u1.y + x.z * u1.z + x.w * u1.w;
        p2 = x.x * u2.x + x.y * u2.y + x.z * u2.z + x.w * u2.w;
        p3 = x.x * u3.x + x.y * u3.y + x.z * u3.z + x.w * u3.w;
    }
#pragma unroll
    for (int m = 1; m < 32; m <<= 1) {
        p0 += __shfl_xor(p0, m, 32);
        p1 += __shfl_xor(p1, m, 32);
        p2 += __shfl_xor(p2, m, 32);
        p3 += __shfl_xor(p3, m, 32);
    }
    if (lane == 0) {
        e_edge[(size_t)e * 4 + 0] = p0;
        e_edge[(size_t)e * 4 + 1] = p1;
        e_edge[(size_t)e * 4 + 2] = p2;
        e_edge[(size_t)e * 4 + 3] = p3;
    }
}

// ---- segment-max pass (both attention paths) -------------------------------
__global__ void max_pass_kernel(const float* __restrict__ e_src, const float* __restrict__ e_dst,
                                const float* __restrict__ e_edge, const int* __restrict__ src,
                                const int* __restrict__ dst, float* __restrict__ m1,
                                float* __restrict__ m2, int n_edges) {
    long idx = (long)blockIdx.x * blockDim.x + threadIdx.x;
    if (idx >= (long)n_edges * 4) return;
    int e = (int)(idx >> 2), h = (int)(idx & 3);
    int s = src[e], d = dst[e];
    float ed = e_dst[d * 4 + h];
    atomicMaxF(&m1[d * 4 + h], lrelu(e_src[s * 4 + h] + ed));
    atomicMaxF(&m2[d * 4 + h], lrelu(e_edge[idx] + ed));
}

// ---- segment-sum pass (recompute logits; stats are L2-resident) ------------
__global__ void sum_pass_kernel(const float* __restrict__ e_src, const float* __restrict__ e_dst,
                                const float* __restrict__ e_edge, const int* __restrict__ src,
                                const int* __restrict__ dst, const float* __restrict__ m1,
                                const float* __restrict__ m2, float* __restrict__ s1,
                                float* __restrict__ s2, int n_edges) {
    long idx = (long)blockIdx.x * blockDim.x + threadIdx.x;
    if (idx >= (long)n_edges * 4) return;
    int e = (int)(idx >> 2), h = (int)(idx & 3);
    int s = src[e], d = dst[e];
    float ed = e_dst[d * 4 + h];
    atomicAdd(&s1[d * 4 + h], __expf(lrelu(e_src[s * 4 + h] + ed) - m1[d * 4 + h]));
    atomicAdd(&s2[d * 4 + h], __expf(lrelu(e_edge[idx] + ed) - m2[d * 4 + h]));
}

// ---- node path: agg[dst] += a1 * h_src[src]  (h_src + agg both L2-hot) -----
__global__ void node_scatter_kernel(const float* __restrict__ h_src, const float* __restrict__ e_src,
                                    const float* __restrict__ e_dst, const float* __restrict__ m1,
                                    const float* __restrict__ s1, const int* __restrict__ src,
                                    const int* __restrict__ dst, float* __restrict__ agg, int n_edges) {
    long idx = (long)blockIdx.x * blockDim.x + threadIdx.x;
    if (idx >= (long)n_edges * 32) return;
    int e = (int)(idx >> 5), g = (int)(idx & 31);
    int h = g >> 3;
    int s = src[e], d = dst[e];
    float el = lrelu(e_src[s * 4 + h] + e_dst[d * 4 + h]);
    float a = __expf(el - m1[d * 4 + h]) / s1[d * 4 + h];
    float4 v = *(const float4*)(h_src + (size_t)s * 128 + g * 4);
    float* ag = agg + (size_t)d * 128 + g * 4;
    atomicAdd(ag + 0, a * v.x);
    atomicAdd(ag + 1, a * v.y);
    atomicAdd(ag + 2, a * v.z);
    atomicAdd(ag + 3, a * v.w);
}

// ---- edge path: recompute h_edge tile with WMMA, scale by a2, scatter ------
// Never materializes h_edge (saves ~1.2 GB of HBM traffic + 410 MB of ws).
__global__ __launch_bounds__(256) void gemm_edge_scatter_kernel(
    const float* __restrict__ Xe, const _Float16* __restrict__ Wh, const float* __restrict__ e_edge,
    const float* __restrict__ e_dst, const float* __restrict__ m2, const float* __restrict__ s2,
    const int* __restrict__ dst, float* __restrict__ agg, int n_rows) {
    __shared__ _Float16 Wlds[128 * 128];
    int tid = threadIdx.x;
    stage_weights_async(Wlds, Wh, tid);

    int wave = tid >> 5, lane = tid & 31;
    int half_ = lane >> 4, lr = lane & 15;
    long r0w = (long)blockIdx.x * 128 + wave * 16;
    long lrow = r0w + lr;
    if (lrow >= n_rows) lrow = n_rows - 1;
    const float* xrow = Xe + lrow * 128;

    v8f acc[8];
#pragma unroll
    for (int nt = 0; nt < 8; ++nt)
#pragma unroll
        for (int i = 0; i < 8; ++i) acc[nt][i] = 0.f;

#pragma unroll
    for (int ks = 0; ks < 4; ++ks) {
        int k0 = ks * 32;
        v16h a = load_a_f32row(xrow, k0, half_);
#pragma unroll
        for (int nt = 0; nt < 8; ++nt) {
            v16h b = load_b_lds(Wlds, nt * 16 + lr, k0, half_);
            acc[nt] = __builtin_amdgcn_wmma_f32_16x16x32_f16(false, a, false, b,
                                                             (short)0, acc[nt], false, false);
        }
    }

    // a2[i][h] for this lane's 8 output rows (edges r0w + half_*8 + i)
    float a2[8][4];
    int dgt[8];
#pragma unroll
    for (int i = 0; i < 8; ++i) {
        long er = r0w + half_ * 8 + i;
        if (er < n_rows) {
            int d = dst[er];
            dgt[i] = d;
#pragma unroll
            for (int h = 0; h < 4; ++h) {
                float el = lrelu(e_edge[er * 4 + h] + e_dst[d * 4 + h]);
                a2[i][h] = __expf(el - m2[d * 4 + h]) / s2[d * 4 + h];
            }
        } else {
            dgt[i] = -1;
#pragma unroll
            for (int h = 0; h < 4; ++h) a2[i][h] = 0.f;
        }
    }
    // column n = nt*16 + lr -> head = nt/2
#pragma unroll
    for (int nt = 0; nt < 8; ++nt) {
        int h = nt >> 1;
        int n = nt * 16 + lr;
#pragma unroll
        for (int i = 0; i < 8; ++i) {
            if (dgt[i] >= 0)
                atomicAdd(&agg[(size_t)dgt[i] * 128 + n], a2[i][h] * acc[nt][i]);
        }
    }
}

// ---- out = h_dst + agg -----------------------------------------------------
__global__ void finalize_kernel(const float* __restrict__ h_dst, const float* __restrict__ agg,
                                float* __restrict__ out, int n_nodes) {
    int idx = blockIdx.x * blockDim.x + threadIdx.x;
    if (idx < n_nodes * 32) {
        float4 a = ((const float4*)h_dst)[idx];
        float4 b = ((const float4*)agg)[idx];
        float4 o;
        o.x = a.x + b.x; o.y = a.y + b.y; o.z = a.z + b.z; o.w = a.w + b.w;
        ((float4*)out)[idx] = o;
    }
}

extern "C" void kernel_launch(void* const* d_in, const int* in_sizes, int n_in,
                              void* d_out, int out_size, void* d_ws, size_t ws_size,
                              hipStream_t stream) {
    const float* X         = (const float*)d_in[0];
    const float* Xe        = (const float*)d_in[1];
    const float* W_src     = (const float*)d_in[2];
    const float* W_dst     = (const float*)d_in[3];
    const float* W_edge    = (const float*)d_in[4];
    const float* attn_src  = (const float*)d_in[5];
    const float* attn_dst  = (const float*)d_in[6];
    const float* attn_edge = (const float*)d_in[7];
    const int*   src       = (const int*)d_in[8];
    const int*   dst       = (const int*)d_in[9];

    const int N = in_sizes[0] / 128;
    const int E = in_sizes[8];

    // f16 weight images first (3 x 16384 halfs = 96 KB, keeps 16B alignment)
    _Float16* Wh_src  = (_Float16*)d_ws;
    _Float16* Wh_dst  = Wh_src + 16384;
    _Float16* Wh_edge = Wh_dst + 16384;
    float* p = (float*)((char*)d_ws + 3 * 16384 * sizeof(_Float16));
    float* h_src  = p; p += (size_t)N * 128;
    float* h_dst  = p; p += (size_t)N * 128;
    float* e_src  = p; p += (size_t)N * 4;
    float* e_dstb = p; p += (size_t)N * 4;
    float* e_edge = p; p += (size_t)E * 4;
    float* U      = p; p += 512;
    float* m1 = p; p += (size_t)N * 4;
    float* m2 = p; p += (size_t)N * 4;
    float* s1 = p; p += (size_t)N * 4;
    float* s2 = p; p += (size_t)N * 4;
    float* agg = p;  // N*128

    init_kernel<<<(N * 128 + 255) / 256, 256, 0, stream>>>(agg, m1, m2, s1, s2, N);

    w_to_f16_kernel<<<16, 256, 0, stream>>>(W_src, Wh_src);
    w_to_f16_kernel<<<16, 256, 0, stream>>>(W_dst, Wh_dst);
    w_to_f16_kernel<<<16, 256, 0, stream>>>(W_edge, Wh_edge);

    gemm_store_kernel<<<(N + 127) / 128, 256, 0, stream>>>(X, Wh_src, h_src, N);
    gemm_store_kernel<<<(N + 127) / 128, 256, 0, stream>>>(X, Wh_dst, h_dst, N);

    e_node_kernel<<<(N + 7) / 8, 256, 0, stream>>>(h_src, h_dst, attn_src, attn_dst, e_src, e_dstb, N);
    u_edge_kernel<<<2, 256, 0, stream>>>(attn_edge, W_edge, U);
    e_edge_kernel<<<(E + 7) / 8, 256, 0, stream>>>(Xe, U, e_edge, E);

    max_pass_kernel<<<(unsigned)(((long)E * 4 + 255) / 256), 256, 0, stream>>>(
        e_src, e_dstb, e_edge, src, dst, m1, m2, E);
    sum_pass_kernel<<<(unsigned)(((long)E * 4 + 255) / 256), 256, 0, stream>>>(
        e_src, e_dstb, e_edge, src, dst, m1, m2, s1, s2, E);

    node_scatter_kernel<<<(unsigned)(((long)E * 32 + 255) / 256), 256, 0, stream>>>(
        h_src, e_src, e_dstb, m1, s1, src, dst, agg, E);
    gemm_edge_scatter_kernel<<<(E + 127) / 128, 256, 0, stream>>>(
        Xe, Wh_edge, e_edge, e_dstb, m2, s2, dst, agg, E);

    finalize_kernel<<<(N * 32 + 255) / 256, 256, 0, stream>>>(h_dst, agg, (float*)d_out, N);
}
